// WindowCrossAttention_69750268887303
// MI455X (gfx1250) — matrix operand order repaired and checked by reading, not compiled, not run on hardware
//
#include <hip/hip_runtime.h>
#include <hip/hip_bf16.h>

typedef __attribute__((ext_vector_type(16))) _Float16 v16h;
typedef __attribute__((ext_vector_type(8)))  float    v8f;

#define NTOK 343
#define NPAD 352
#define DIMC 192
#define NH   6
#define HD   32
#define NB   512
#define NWIN 64
#define RT   22    // row tiles of 16 over 352
#define KTPV 11    // K tiles of 32 over 352
#define KSQ  6     // K steps of 32 over 192

static __device__ __forceinline__ v8f wmma16(v16h a, v16h b, v8f c) {
  return __builtin_amdgcn_wmma_f32_16x16x32_f16(false, a, false, b, (short)0, c, false, false);
}

// A fragment: M = lane%16; halves 0..15 = K 2*dA..2*dA+15 (contiguous)
template <bool F16>
static __device__ __forceinline__ v16h load_a_frag(const void* base, size_t elemOff, int dA) {
  v16h a;
  if (F16) {
    const _Float16* p = (const _Float16*)base + elemOff;
#pragma unroll
    for (int i = 0; i < 16; i++) a[i] = p[2 * dA + i];
  } else {
    const float* p = (const float*)base + elemOff;
#pragma unroll
    for (int i = 0; i < 16; i++) a[i] = (_Float16)p[2 * dA + i];
  }
  return a;
}

// B fragment: N = lane%16; halves = 16 contiguous K values (elemOff includes dB)
template <bool F16>
static __device__ __forceinline__ v16h load_b_frag(const void* base, size_t elemOff) {
  v16h bf;
  if (F16) {
    const _Float16* p = (const _Float16*)base + elemOff;
#pragma unroll
    for (int i = 0; i < 16; i++) bf[i] = p[i];
  } else {
    const float* p = (const float*)base + elemOff;
#pragma unroll
    for (int i = 0; i < 16; i++) bf[i] = (_Float16)p[i];
  }
  return bf;
}

// ---------------- elementwise f32 -> f16 (one-time pre-conversion) --------
__global__ void cvt_kernel(const float* __restrict__ in, _Float16* __restrict__ out, int n) {
  int i = (blockIdx.x * blockDim.x + threadIdx.x) * 4;
  if (i + 3 < n) {
    float4 v = *(const float4*)(in + i);
    out[i + 0] = (_Float16)v.x;
    out[i + 1] = (_Float16)v.y;
    out[i + 2] = (_Float16)v.z;
    out[i + 3] = (_Float16)v.w;
  } else {
    for (int j = i; j < n; j++) out[j] = (_Float16)in[j];
  }
}

// ---------------- kernel 1: CPB MLP -> bias_table[2197][6] ----------------
__global__ void cpb_kernel(const float* __restrict__ rel, const float* __restrict__ w1,
                           const float* __restrict__ b1, const float* __restrict__ w2,
                           float* __restrict__ bias_table) {
  int i = blockIdx.x * blockDim.x + threadIdx.x;
  if (i >= 2197) return;
  float r0 = rel[i * 3 + 0], r1 = rel[i * 3 + 1], r2 = rel[i * 3 + 2];
  float acc[NH];
#pragma unroll
  for (int h = 0; h < NH; h++) acc[h] = 0.f;
  for (int j = 0; j < 512; j++) {
    float hv = fmaf(w1[j * 3 + 0], r0, fmaf(w1[j * 3 + 1], r1, fmaf(w1[j * 3 + 2], r2, b1[j])));
    hv = fmaxf(hv, 0.f);
#pragma unroll
    for (int h = 0; h < NH; h++) acc[h] = fmaf(w2[h * 512 + j], hv, acc[h]);
  }
#pragma unroll
  for (int h = 0; h < NH; h++) bias_table[i * NH + h] = acc[h];
}

// ---------------- kernel 2: gather + 16*sigmoid -> rpb_sig[H][N][N] -------
__global__ void rpb_kernel(const float* __restrict__ bias_table, const int* __restrict__ rp_index,
                           float* __restrict__ rpb_sig) {
  int idx = blockIdx.x * blockDim.x + threadIdx.x;
  const int total = NH * NTOK * NTOK;
  if (idx >= total) return;
  int h  = idx / (NTOK * NTOK);
  int mn = idx % (NTOK * NTOK);
  float v = bias_table[rp_index[mn] * NH + h];
  rpb_sig[idx] = 16.f / (1.f + __expf(-v));
}

// ---------------- kernel 3: fused window cross-attention ------------------
// one workgroup (256 thr = 8 wave32) per (b, h); F16 => pre-converted inputs
template <bool F16>
__global__ __launch_bounds__(256) void attn_kernel(
    const void* __restrict__ xsrc, const void* __restrict__ ysrc,
    const float* __restrict__ mask,
    const void* __restrict__ qkv1w, const void* __restrict__ qkv2w,
    const float* __restrict__ q2_bias, const float* __restrict__ v1_bias,
    const float* __restrict__ logit_scale,
    const float* __restrict__ rpb_sig,
    _Float16* __restrict__ cross) {
  extern __shared__ char smem[];
  _Float16* q_a  = (_Float16*)(smem + 0);        // [352][32]  f16
  _Float16* k_b  = (_Float16*)(smem + 22528);    // [352][32]  f16
  _Float16* vT   = (_Float16*)(smem + 45056);    // [32][352]  f16
  float*    Pf   = (float*)   (smem + 67584);    // [16][352]  f32
  _Float16* Ph   = (_Float16*)(smem + 90112);    // [16][352]  f16
  float*    Op   = (float*)   (smem + 101376);   // [8][16][16] f32
  float*    red  = (float*)   (smem + 109568);   // [16][16]
  float*    rstat= (float*)   (smem + 110592);   // [16]

  const int tid  = threadIdx.x;
  const int lane = tid & 31;
  const int wave = tid >> 5;
  const int bh = blockIdx.x;
  const int b  = bh / NH;
  const int h  = bh % NH;
  const int w  = b & (NWIN - 1);

  const int l16 = lane & 15;
  const int hiL = (lane >> 4) & 1;
  const int dA  = hiL * 8;    // A-fragment K chunk offset
  const int dB  = hiL * 16;   // B-fragment K offset
  const int r8  = hiL * 8;    // C-fragment row offset

  // ---- phase 1: per-head q/k/v GEMMs into LDS (f16) ----
  for (int t = wave; t < 132; t += 8) {
    const int tensor = t / 44;      // 0=q (from y), 1=k, 2=v (from x)
    const int tt = t % 44;
    const int rt = tt >> 1, nt = tt & 1;
    const void* src;
    const void* wsrc;
    size_t wrow;
    float bias;
    if (tensor == 0) {
      src = ysrc; wsrc = qkv2w;
      wrow = (size_t)(h * HD + nt * 16 + l16) * DIMC;
      bias = q2_bias[h * HD + nt * 16 + l16];
    } else if (tensor == 1) {
      src = xsrc; wsrc = qkv1w;
      wrow = (size_t)(DIMC + h * HD + nt * 16 + l16) * DIMC;
      bias = 0.f;
    } else {
      src = xsrc; wsrc = qkv1w;
      wrow = (size_t)(2 * DIMC + h * HD + nt * 16 + l16) * DIMC;
      bias = v1_bias[h * HD + nt * 16 + l16];
    }
    int tokA = rt * 16 + l16;
    if (tokA > NTOK - 1) tokA = NTOK - 1;             // clamp, masked later
    const size_t arow = ((size_t)b * NTOK + tokA) * DIMC;
    v8f c;
#pragma unroll
    for (int r = 0; r < 8; r++) c[r] = bias;
#pragma unroll
    for (int ks = 0; ks < KSQ; ks++) {
      v16h a  = load_a_frag<F16>(src,  arow + ks * 32, dA);
      v16h bf = load_b_frag<F16>(wsrc, wrow + ks * 32 + dB);
      c = wmma16(a, bf, c);
    }
    if (tensor == 2) {
      _Float16* dst = vT + (size_t)(nt * 16 + l16) * NPAD + rt * 16 + r8;
#pragma unroll
      for (int r = 0; r < 8; r++) dst[r] = (_Float16)c[r];
    } else {
      _Float16* dst = (tensor == 0) ? q_a : k_b;
#pragma unroll
      for (int r = 0; r < 8; r++)
        dst[(size_t)(rt * 16 + r + r8) * HD + nt * 16 + l16] = (_Float16)c[r];
    }
  }
  __syncthreads();

  // ---- phase 2: cosine-normalize rows; fold logit scale into q ----
  const float lsc = __expf(fminf(logit_scale[h], 4.6051702f)); // log(100)
  for (int row = tid; row < NTOK; row += 256) {
    float ssq = 0.f;
#pragma unroll
    for (int d = 0; d < HD; d++) { float v = (float)q_a[row * HD + d]; ssq = fmaf(v, v, ssq); }
    float inv = lsc / fmaxf(sqrtf(ssq), 1e-12f);
#pragma unroll
    for (int d = 0; d < HD; d++) q_a[row * HD + d] = (_Float16)((float)q_a[row * HD + d] * inv);
    ssq = 0.f;
#pragma unroll
    for (int d = 0; d < HD; d++) { float v = (float)k_b[row * HD + d]; ssq = fmaf(v, v, ssq); }
    inv = 1.f / fmaxf(sqrtf(ssq), 1e-12f);
#pragma unroll
    for (int d = 0; d < HD; d++) k_b[row * HD + d] = (_Float16)((float)k_b[row * HD + d] * inv);
  }
  __syncthreads();

  const float* rpbH  = rpb_sig + (size_t)h * NTOK * NTOK;
  const float* maskW = mask    + (size_t)w * NTOK * NTOK;

  // ---- phase 3: 16-row blocks: S -> softmax -> PV -> store ----
  for (int rb = 0; rb < RT; rb++) {
    // (a) S = q kT, one WMMA per 16x16 tile; +rpb +mask epilogue
    for (int ct = wave; ct < RT; ct += 8) {
      v16h a  = load_a_frag<true>(q_a, (size_t)(rb * 16 + l16) * HD, dA);
      v16h bf = load_b_frag<true>(k_b, (size_t)(ct * 16 + l16) * HD + dB);
      v8f c = {};
      c = wmma16(a, bf, c);
      const int n = ct * 16 + l16;
#pragma unroll
      for (int r = 0; r < 8; r++) {
        const int lr = r + r8;
        const int m = rb * 16 + lr;
        float val;
        if (n >= NTOK)      val = -1e30f;
        else if (m >= NTOK) val = 0.f;
        else val = c[r] + rpbH[(size_t)m * NTOK + n] + maskW[(size_t)m * NTOK + n];
        Pf[lr * NPAD + n] = val;
      }
    }
    __syncthreads();

    // (b) softmax: row max, then exp -> f16 directly; 1/sum deferred to (d)
    {
      const int row = tid & 15, slot = tid >> 4;
      const float* prow = Pf + row * NPAD + slot * 22;
      float mx = -1e30f;
#pragma unroll
      for (int j = 0; j < 22; j++) mx = fmaxf(mx, prow[j]);
      red[row * 16 + slot] = mx;
      __syncthreads();
      if (tid < 16) {
        float m2 = -1e30f;
        for (int s = 0; s < 16; s++) m2 = fmaxf(m2, red[tid * 16 + s]);
        rstat[tid] = m2;
      }
      __syncthreads();
      const float rm = rstat[row];
      _Float16* ph = Ph + row * NPAD + slot * 22;
      float s = 0.f;
#pragma unroll
      for (int j = 0; j < 22; j++) {
        float e = __expf(prow[j] - rm);
        s += e;
        ph[j] = (_Float16)e;
      }
      red[row * 16 + slot] = s;
      __syncthreads();
      if (tid < 16) {
        float s2 = 0.f;
        for (int s3 = 0; s3 < 16; s3++) s2 += red[tid * 16 + s3];
        rstat[tid] = 1.f / s2;     // read later in (d)
      }
    }
    __syncthreads();

    // (c) PV: O[16,32], K split over wave pairs, partial accum per wave
    {
      const int nt = wave & 1;
      v8f c = {};
      for (int kt = (wave >> 1); kt < KTPV; kt += 4) {
        v16h a  = load_a_frag<true>(Ph, (size_t)l16 * NPAD + kt * 32, dA);
        v16h bf = load_b_frag<true>(vT, (size_t)(nt * 16 + l16) * NPAD + kt * 32 + dB);
        c = wmma16(a, bf, c);
      }
#pragma unroll
      for (int r = 0; r < 8; r++) Op[wave * 256 + (r + r8) * 16 + l16] = c[r];
    }
    __syncthreads();

    // (d) reduce 4 partials per half, apply 1/rowsum, store f16
#pragma unroll
    for (int i = 0; i < 2; i++) {
      const int idx = tid + 256 * i;          // 0..511
      const int row = idx >> 5, d = idx & 31;
      const int nt = d >> 4, dc = d & 15;
      float s = Op[(0 + nt) * 256 + row * 16 + dc] + Op[(2 + nt) * 256 + row * 16 + dc] +
                Op[(4 + nt) * 256 + row * 16 + dc] + Op[(6 + nt) * 256 + row * 16 + dc];
      s *= rstat[row];
      const int token = rb * 16 + row;
      if (token < NTOK)
        cross[((size_t)b * NTOK + token) * DIMC + h * HD + d] = (_Float16)s;
    }
    __syncthreads();
  }
}

// ---------------- kernel 4: output projection GEMM ------------------------
// out[M,192] = cross[M,192] @ proj_w[192,192]^T + proj_b ; M = 175616 (exact tiles)
template <bool F16>
__global__ __launch_bounds__(256) void proj_kernel(
    const _Float16* __restrict__ cross, const void* __restrict__ proj_w,
    const float* __restrict__ proj_b, float* __restrict__ out) {
  const int tid = threadIdx.x;
  const int lane = tid & 31;
  const int wave = tid >> 5;
  const int l16 = lane & 15;
  const int hiL = (lane >> 4) & 1;
  const int dA = hiL * 8, dB = hiL * 16, r8 = hiL * 8;
  const int mt = blockIdx.x * 8 + wave;

  v16h A[KSQ];
#pragma unroll
  for (int ks = 0; ks < KSQ; ks++)
    A[ks] = load_a_frag<true>(cross, (size_t)(mt * 16 + l16) * DIMC + ks * 32, dA);

  for (int ct = 0; ct < 12; ct++) {
    const float bias = proj_b[ct * 16 + l16];
    v8f c;
#pragma unroll
    for (int r = 0; r < 8; r++) c[r] = bias;
    const size_t wrow = (size_t)(ct * 16 + l16) * DIMC;
#pragma unroll
    for (int ks = 0; ks < KSQ; ks++) {
      v16h bf = load_b_frag<F16>(proj_w, wrow + ks * 32 + dB);
      c = wmma16(A[ks], bf, c);
    }
#pragma unroll
    for (int r = 0; r < 8; r++)
      out[(size_t)(mt * 16 + r + r8) * DIMC + ct * 16 + l16] = c[r];
  }
}

// ---------------------------------------------------------------------------
extern "C" void kernel_launch(void* const* d_in, const int* in_sizes, int n_in,
                              void* d_out, int out_size, void* d_ws, size_t ws_size,
                              hipStream_t stream) {
  const float* x           = (const float*)d_in[0];
  const float* y           = (const float*)d_in[1];
  const float* mask        = (const float*)d_in[2];
  const float* qkv1_w      = (const float*)d_in[3];
  const float* qkv2_w      = (const float*)d_in[4];
  // d_in[5] = q1_bias (unused: k bias is the zero middle slice of qkv1_b)
  const float* v1_bias     = (const float*)d_in[6];
  const float* q2_bias     = (const float*)d_in[7];
  // d_in[8] = v2_bias (unused: only q taken from qkv2)
  const float* logit_scale = (const float*)d_in[9];
  const float* cpb_w1      = (const float*)d_in[10];
  const float* cpb_b1      = (const float*)d_in[11];
  const float* cpb_w2      = (const float*)d_in[12];
  const float* proj_w      = (const float*)d_in[13];
  const float* proj_b      = (const float*)d_in[14];
  const float* rel_table   = (const float*)d_in[15];
  const int*   rp_index    = (const int*)d_in[16];

  // workspace layout (64B aligned)
  char* ws = (char*)d_ws;
  float*    bias_table = (float*)ws;                      // 52,728 B
  float*    rpb_sig    = (float*)(ws + 52736);            // 2,823,576 B
  _Float16* cross      = (_Float16*)(ws + 2876352);       // 67,436,544 B -> end 70,312,896
  _Float16* qkv1_16    = (_Float16*)(ws + 70312896);      // 221,184 B
  _Float16* qkv2_16    = (_Float16*)(ws + 70534080);      // 221,184 B
  _Float16* projw_16   = (_Float16*)(ws + 70755264);      //  73,728 B
  _Float16* x16        = (_Float16*)(ws + 70828992);      // 67,436,544 B
  _Float16* y16        = (_Float16*)(ws + 138265536);     // 67,436,544 B -> end 205,702,080
  const bool fast = (ws_size >= (size_t)205702080ull);

  const size_t SMEM = 110656;
  (void)hipFuncSetAttribute((const void*)attn_kernel<true>,
                            hipFuncAttributeMaxDynamicSharedMemorySize, (int)SMEM);
  (void)hipFuncSetAttribute((const void*)attn_kernel<false>,
                            hipFuncAttributeMaxDynamicSharedMemorySize, (int)SMEM);

  cpb_kernel<<<(2197 + 127) / 128, 128, 0, stream>>>(rel_table, cpb_w1, cpb_b1, cpb_w2, bias_table);

  const int rpb_total = NH * NTOK * NTOK;
  rpb_kernel<<<(rpb_total + 255) / 256, 256, 0, stream>>>(bias_table, rp_index, rpb_sig);

  if (fast) {
    const int nxy = NB * NTOK * DIMC;        // 33,718,272 (div by 4)
    cvt_kernel<<<(nxy / 4 + 255) / 256, 256, 0, stream>>>(x, x16, nxy);
    cvt_kernel<<<(nxy / 4 + 255) / 256, 256, 0, stream>>>(y, y16, nxy);
    const int nw = 3 * DIMC * DIMC;          // 110,592
    cvt_kernel<<<(nw / 4 + 255) / 256, 256, 0, stream>>>(qkv1_w, qkv1_16, nw);
    cvt_kernel<<<(nw / 4 + 255) / 256, 256, 0, stream>>>(qkv2_w, qkv2_16, nw);
    const int np = DIMC * DIMC;              // 36,864
    cvt_kernel<<<(np / 4 + 255) / 256, 256, 0, stream>>>(proj_w, projw_16, np);

    attn_kernel<true><<<NB * NH, 256, SMEM, stream>>>(
        x16, y16, mask, qkv1_16, qkv2_16, q2_bias, v1_bias, logit_scale, rpb_sig, cross);
    proj_kernel<true><<<(NB * NTOK) / (16 * 8), 256, 0, stream>>>(cross, projw_16, proj_b,
                                                                  (float*)d_out);
  } else {
    attn_kernel<false><<<NB * NH, 256, SMEM, stream>>>(
        x, y, mask, qkv1_w, qkv2_w, q2_bias, v1_bias, logit_scale, rpb_sig, cross);
    proj_kernel<false><<<(NB * NTOK) / (16 * 8), 256, 0, stream>>>(cross, proj_w, proj_b,
                                                                   (float*)d_out);
  }
}
